// SDGraphEncoder_32109175505368
// MI455X (gfx1250) — matrix operand order repaired and checked by reading, not compiled, not run on hardware
//
#include <hip/hip_runtime.h>
#include <hip/hip_bf16.h>
#include <math.h>

typedef __attribute__((ext_vector_type(16))) __bf16 v16bf;
typedef __attribute__((ext_vector_type(8)))  float  v8f;

#define BN_EPS 1e-5f

__device__ __forceinline__ float gelu_f(float x) {
    return 0.5f * x * (1.0f + erff(x * 0.7071067811865475f));
}
__device__ __forceinline__ float bn_f(float x, const float* p, int C, int c) {
    float g = p[c], b = p[C + c], m = p[2 * C + c], v = p[3 * C + c];
    return (x - m) * (g * rsqrtf(v + BN_EPS)) + b;
}

// ---- WMMA fragment loaders (contiguous k order for A and B, 16x16x32) ----
// A (16xK tile, row-major in LDS, stride lda halfs):
//   lane<16:  M=lane,    halves 0..15 -> K=h
//   lane>=16: M=lane-16, halves 0..15 -> K=h+16
__device__ __forceinline__ v16bf load_frag_a(const __bf16* a, int lda) {
    int lane = threadIdx.x & 31;
    int m = lane & 15;
    int kb = (lane >> 4) << 4;       // 0 or 16
    const __bf16* row = a + m * lda;
    v16bf f;
#pragma unroll
    for (int h = 0; h < 16; ++h) {
        int k = h + kb;
        f[h] = row[k];
    }
    return f;
}
// B (Kx16 tile) stored transposed in LDS as Bt[N][K] (stride ldb halfs):
//   lane<16: N=lane, K=h ; lane>=16: N=lane-16, K=h+16
__device__ __forceinline__ v16bf load_frag_b(const __bf16* bt, int ldb) {
    int lane = threadIdx.x & 31;
    int n = lane & 15;
    int kb = (lane >> 4) << 4;       // 0 or 16
    const __bf16* row = bt + n * ldb + kb;
    v16bf f;
#pragma unroll
    for (int h = 0; h < 16; ++h) f[h] = row[h];
    return f;
}
// C/D f32: VGPR r -> M = r + (lane>=16 ? 8 : 0), N = lane & 15

// ============ DenseToSparse: conv (1,2) 64->90, BN, GELU ============
__global__ void k_dts1(const float* __restrict__ dense, const float* __restrict__ w,
                       const float* __restrict__ bias, const float* __restrict__ bn,
                       float* __restrict__ out) {
    int t = blockIdx.x * blockDim.x + threadIdx.x;
    const int TOT = 16 * 90 * 32 * 63;
    if (t >= TOT) return;
    int wi = t % 63;
    int s  = (t / 63) % 32;
    int o  = (t / (63 * 32)) % 90;
    int b  = t / (63 * 32 * 90);
    float acc = bias[o];
    const float* dp = dense + (size_t)(b * 64) * 2048 + s * 64 + wi;
    const float* wp = w + o * 128;
#pragma unroll 8
    for (int c = 0; c < 64; ++c)
        acc += wp[2 * c] * dp[c * 2048] + wp[2 * c + 1] * dp[c * 2048 + 1];
    out[t] = gelu_f(bn_f(acc, bn, 90, o));
}

// ============ DenseToSparse: conv (1,2) 90->128, BN, GELU, max over width ============
__global__ void k_dts2max(const float* __restrict__ h1, const float* __restrict__ w,
                          const float* __restrict__ bias, const float* __restrict__ bn,
                          float* __restrict__ out) {
    int t = blockIdx.x * blockDim.x + threadIdx.x;
    const int TOT = 16 * 128 * 32;
    if (t >= TOT) return;
    int s = t % 32;
    int o = (t / 32) % 128;
    int b = t / (32 * 128);
    float best = -1e30f;
    const float* wp = w + o * 180;
    for (int wi = 0; wi < 62; ++wi) {
        float acc = bias[o];
        const float* hp = h1 + ((size_t)(b * 90) * 32 + s) * 63 + wi;
        for (int c = 0; c < 90; ++c)
            acc += wp[2 * c] * hp[c * 32 * 63] + wp[2 * c + 1] * hp[c * 32 * 63 + 1];
        best = fmaxf(best, gelu_f(bn_f(acc, bn, 128, o)));
    }
    out[t] = best;
}

// ============ build union_sparse [16,256,32] ============
__global__ void k_build_usp(const float* __restrict__ sparse, const float* __restrict__ spdn,
                            float* __restrict__ usp) {
    int t = blockIdx.x * blockDim.x + threadIdx.x;
    if (t >= 16 * 256 * 32) return;
    int i = t % 32;
    int c = (t / 32) % 256;
    int b = t / (32 * 256);
    usp[t] = (c < 128) ? sparse[(size_t)(b * 128 + c) * 32 + i]
                       : spdn[(size_t)(b * 128 + (c - 128)) * 32 + i];
}

// ============ build union_dense f32 [16,192,2048] + bf16 copy ============
__global__ void k_build_uden(const float* __restrict__ dense, const float* __restrict__ sparse,
                             float* __restrict__ uden, __bf16* __restrict__ xbf) {
    int t = blockIdx.x * blockDim.x + threadIdx.x;
    if (t >= 16 * 192 * 2048) return;
    int p = t % 2048;
    int c = (t / 2048) % 192;
    int b = t / (2048 * 192);
    float v = (c < 64) ? dense[(size_t)(b * 64 + c) * 2048 + p]
                       : sparse[(size_t)(b * 128 + (c - 64)) * 32 + (p >> 6)];
    uden[t] = v;
    xbf[t]  = (__bf16)v;
}

// ============ squared norms for dense knn ============
__global__ void k_sq(const float* __restrict__ uden, float* __restrict__ sqn) {
    int t = blockIdx.x * blockDim.x + threadIdx.x;
    if (t >= 16 * 2048) return;
    int p = t % 2048, b = t / 2048;
    float s = 0.f;
    for (int k = 0; k < 192; ++k) {
        float x = uden[((size_t)b * 192 + k) * 2048 + p];
        s += x * x;
    }
    sqn[t] = s;
}

// ============ sparse knn (n=32, k=2) ============
__global__ void k_spknn(const float* __restrict__ usp, int* __restrict__ spknn) {
    int b = blockIdx.x, tid = threadIdx.x;
    __shared__ float sq[32];
    __shared__ float dm[32 * 32];
    if (tid < 32) {
        float s = 0.f;
        for (int c = 0; c < 256; ++c) {
            float x = usp[(size_t)(b * 256 + c) * 32 + tid];
            s += x * x;
        }
        sq[tid] = s;
    }
    __syncthreads();
    for (int pid = tid; pid < 1024; pid += 256) {
        int i = pid >> 5, j = pid & 31;
        float dot = 0.f;
        for (int c = 0; c < 256; ++c)
            dot += usp[(size_t)(b * 256 + c) * 32 + i] * usp[(size_t)(b * 256 + c) * 32 + j];
        dm[pid] = sq[i] + sq[j] - 2.f * dot;
    }
    __syncthreads();
    if (tid < 32) {
        float m0 = 1e30f, m1 = 1e30f;
        int i0 = 0, i1 = 0;
        for (int j = 0; j < 32; ++j) {
            float v = dm[tid * 32 + j];
            if (v < m0) { m1 = m0; i1 = i0; m0 = v; i0 = j; }
            else if (v < m1) { m1 = v; i1 = j; }
        }
        spknn[(b * 32 + tid) * 2]     = i0;
        spknn[(b * 32 + tid) * 2 + 1] = i1;
    }
}

// ============ sparse DGCNN: edge 512 -> 256 -> 128, max over k=2 ============
__global__ void k_spdgcnn(const float* __restrict__ usp, const int* __restrict__ spknn,
                          const float* __restrict__ w1, const float* __restrict__ b1,
                          const float* __restrict__ bn1,
                          const float* __restrict__ w2, const float* __restrict__ b2,
                          const float* __restrict__ bn2, float* __restrict__ out) {
    int i = blockIdx.x, b = blockIdx.y, tid = threadIdx.x;  // block 256
    __shared__ float sxi[256];
    __shared__ float se[512];
    __shared__ float sy1[256];
    sxi[tid] = usp[(size_t)(b * 256 + tid) * 32 + i];
    __syncthreads();
    float best = -1e30f;
    for (int s = 0; s < 2; ++s) {
        int j = spknn[(b * 32 + i) * 2 + s];
        float xj = usp[(size_t)(b * 256 + tid) * 32 + j];
        se[tid]       = xj - sxi[tid];
        se[256 + tid] = sxi[tid];
        __syncthreads();
        {
            float acc = b1[tid];
            const float* wr = w1 + (size_t)tid * 512;
            for (int c = 0; c < 512; ++c) acc += wr[c] * se[c];
            sy1[tid] = gelu_f(bn_f(acc, bn1, 256, tid));
        }
        __syncthreads();
        if (tid < 128) {
            float acc = b2[tid];
            const float* wr = w2 + (size_t)tid * 256;
            for (int c = 0; c < 256; ++c) acc += wr[c] * sy1[c];
            best = fmaxf(best, gelu_f(bn_f(acc, bn2, 128, tid)));
        }
        __syncthreads();
    }
    if (tid < 128) out[(size_t)(b * 128 + tid) * 32 + i] = best;
}

// ============ Gram + distance tile (WMMA), one batch per launch ============
// grid (32 jtiles, 128 itiles), block 128 (4 waves)
__global__ void k_gram(const __bf16* __restrict__ xbf, const float* __restrict__ sqn,
                       float* __restrict__ dist, int b) {
    __shared__ __bf16 A[16 * 192];
    __shared__ __bf16 B[64 * 192];
    int i0 = blockIdx.y * 16, j0 = blockIdx.x * 64;
    const __bf16* X = xbf + (size_t)b * 192 * 2048;
    int tid = threadIdx.x;
    for (int e = tid; e < 16 * 192; e += 128) {
        int m = e / 192, k = e % 192;
        A[m * 192 + k] = X[(size_t)k * 2048 + i0 + m];
    }
    for (int e = tid; e < 64 * 192; e += 128) {
        int n = e / 192, k = e % 192;
        B[n * 192 + k] = X[(size_t)k * 2048 + j0 + n];
    }
    __syncthreads();
    int w = tid >> 5;
    v8f c = {};
#pragma unroll
    for (int kk = 0; kk < 6; ++kk) {
        v16bf a  = load_frag_a(A + kk * 32, 192);
        v16bf bb = load_frag_b(B + (w * 16) * 192 + kk * 32, 192);
        c = __builtin_amdgcn_wmma_f32_16x16x32_bf16(false, a, false, bb, (short)0, c, false, false);
    }
    int lane = tid & 31;
    int n = lane & 15, mh = (lane >> 4) * 8;
    const float* sqb = sqn + b * 2048;
#pragma unroll
    for (int r = 0; r < 8; ++r) {
        int i = i0 + r + mh;
        int j = j0 + w * 16 + n;
        dist[(size_t)i * 2048 + j] = sqb[i] + sqb[j] - 2.0f * c[r];
    }
}

// ============ row-wise top-10 (smallest), one wave per row ============
__global__ void k_topk(const float* __restrict__ dist, int* __restrict__ knn, int b) {
    int wv = threadIdx.x >> 5, lane = threadIdx.x & 31;
    int row = blockIdx.x * 8 + wv;
    float best[10]; int bidx[10];
#pragma unroll
    for (int t = 0; t < 10; ++t) { best[t] = 1e30f; bidx[t] = 0; }
    const float* dr = dist + (size_t)row * 2048;
    for (int j = lane; j < 2048; j += 32) {
        float v = dr[j];
        if (v < best[9]) {
            int pos = 9;
            while (pos > 0 && best[pos - 1] > v) {
                best[pos] = best[pos - 1]; bidx[pos] = bidx[pos - 1]; --pos;
            }
            best[pos] = v; bidx[pos] = j;
        }
    }
    __shared__ float sv[8][320];
    __shared__ int   si[8][320];
#pragma unroll
    for (int t = 0; t < 10; ++t) { sv[wv][lane * 10 + t] = best[t]; si[wv][lane * 10 + t] = bidx[t]; }
    __syncthreads();
    if (lane == 0) {
        int* o = knn + ((size_t)b * 2048 + row) * 10;
        for (int t = 0; t < 10; ++t) {
            float mv = 1e30f; int mi = 0, mp = 0;
            for (int q = 0; q < 320; ++q)
                if (sv[wv][q] < mv) { mv = sv[wv][q]; mi = si[wv][q]; mp = q; }
            sv[wv][mp] = 1e31f;
            o[t] = mi;
        }
    }
}

// ============ P/Q GEMM (WMMA): [156x192] @ [192x2048] per batch ============
// grid (32 ntiles, 10 mtiles, 16 batches), block 128
__global__ void k_pq(const float* __restrict__ W, const __bf16* __restrict__ xbf,
                     float* __restrict__ out, int qmode) {
    __shared__ __bf16 A[16 * 192];
    __shared__ __bf16 B[64 * 192];
    int b = blockIdx.z, m0 = blockIdx.y * 16, j0 = blockIdx.x * 64;
    int tid = threadIdx.x;
    for (int e = tid; e < 16 * 192; e += 128) {
        int m = e / 192, k = e % 192;
        int co = m0 + m;
        float wv = 0.f;
        if (co < 156)
            wv = qmode ? (W[(size_t)co * 384 + 192 + k] - W[(size_t)co * 384 + k])
                       : W[(size_t)co * 384 + k];
        A[m * 192 + k] = (__bf16)wv;
    }
    const __bf16* X = xbf + (size_t)b * 192 * 2048;
    for (int e = tid; e < 64 * 192; e += 128) {
        int n = e / 192, k = e % 192;
        B[n * 192 + k] = X[(size_t)k * 2048 + j0 + n];
    }
    __syncthreads();
    int w = tid >> 5;
    v8f c = {};
#pragma unroll
    for (int kk = 0; kk < 6; ++kk) {
        v16bf a  = load_frag_a(A + kk * 32, 192);
        v16bf bb = load_frag_b(B + (w * 16) * 192 + kk * 32, 192);
        c = __builtin_amdgcn_wmma_f32_16x16x32_bf16(false, a, false, bb, (short)0, c, false, false);
    }
    int lane = tid & 31;
    int n = lane & 15, mh = (lane >> 4) * 8;
#pragma unroll
    for (int r = 0; r < 8; ++r) {
        int co = m0 + r + mh;
        if (co < 156)
            out[((size_t)b * 156 + co) * 2048 + j0 + w * 16 + n] = c[r];
    }
}

// ============ fused dense conv2 (WMMA): gather + BN1+GELU -> 64x160 GEMM ============
// -> BN2+GELU -> max over 10 neighbor slots.  grid (128 ptiles, 16 batches), block 128
__global__ void k_dnconv2(const float* __restrict__ P, const float* __restrict__ Q,
                          const int* __restrict__ knn, const float* __restrict__ W2,
                          const float* __restrict__ b1, const float* __restrict__ bn1,
                          const float* __restrict__ b2, const float* __restrict__ bn2,
                          float* __restrict__ out) {
    __shared__ __bf16 Wl[64 * 160];
    __shared__ __bf16 El[80 * 160];   // 5 neighbor slots x 16 points per phase
    __shared__ int   jc[160];
    __shared__ float s1[160], sh1[160];
    int b = blockIdx.y, tile = blockIdx.x, tid = threadIdx.x;
    int p0 = tile * 16;
    for (int e = tid; e < 64 * 160; e += 128) {
        int m = e / 160, k = e % 160;
        Wl[e] = (__bf16)(k < 156 ? W2[(size_t)m * 156 + k] : 0.f);
    }
    for (int k = tid; k < 160; k += 128) {
        if (k < 156) {
            float g = bn1[k], be = bn1[156 + k], mm = bn1[312 + k], vv = bn1[468 + k];
            float s = g * rsqrtf(vv + BN_EPS);
            s1[k] = s; sh1[k] = be + s * (b1[k] - mm);
        } else { s1[k] = 0.f; sh1[k] = 0.f; }
    }
    for (int cidx = tid; cidx < 160; cidx += 128) {
        int s = cidx >> 4, p = cidx & 15;
        jc[cidx] = knn[((size_t)b * 2048 + p0 + p) * 10 + s];
    }
    __syncthreads();
    const float* Pb = P + (size_t)b * 156 * 2048;
    const float* Qb = Q + (size_t)b * 156 * 2048;
    int w = tid >> 5, lane = tid & 31, n = lane & 15, mh = (lane >> 4) * 8;
    float s2v[8], sh2v[8], best[8];
#pragma unroll
    for (int r = 0; r < 8; ++r) {
        int co = w * 16 + r + mh;
        float g = bn2[co], be = bn2[64 + co], mm = bn2[128 + co], vv = bn2[192 + co];
        float s = g * rsqrtf(vv + BN_EPS);
        s2v[r] = s; sh2v[r] = be + s * (b2[co] - mm);
        best[r] = -1e30f;
    }
    for (int half = 0; half < 2; ++half) {
        for (int e = tid; e < 80 * 160; e += 128) {
            int cc = e / 160, k = e % 160;
            float v = 0.f;
            if (k < 156) {
                int sl = cc >> 4, p = cc & 15;
                int gcol = (half * 5 + sl) * 16 + p;
                int j = jc[gcol];
                float x = Pb[(size_t)k * 2048 + j] + Qb[(size_t)k * 2048 + p0 + p];
                v = gelu_f(s1[k] * x + sh1[k]);
            }
            El[cc * 160 + k] = (__bf16)v;
        }
        __syncthreads();
        for (int sl = 0; sl < 5; ++sl) {
            v8f c = {};
#pragma unroll
            for (int kk = 0; kk < 5; ++kk) {
                v16bf a  = load_frag_a(Wl + (w * 16) * 160 + kk * 32, 160);
                v16bf bb = load_frag_b(El + (sl * 16) * 160 + kk * 32, 160);
                c = __builtin_amdgcn_wmma_f32_16x16x32_bf16(false, a, false, bb, (short)0, c, false, false);
            }
#pragma unroll
            for (int r = 0; r < 8; ++r)
                best[r] = fmaxf(best[r], gelu_f(s2v[r] * c[r] + sh2v[r]));
        }
        __syncthreads();
    }
#pragma unroll
    for (int r = 0; r < 8; ++r) {
        int co = w * 16 + r + mh;
        out[((size_t)b * 64 + co) * 2048 + p0 + n] = best[r];
    }
}

// ============ down_sample: (1,3) conv stride (1,2) pad (0,1), BN, GELU ============
__global__ void k_down(const float* __restrict__ dnf, const float* __restrict__ w,
                       const float* __restrict__ bias, const float* __restrict__ bn,
                       float* __restrict__ out) {
    int t = blockIdx.x * blockDim.x + threadIdx.x;
    if (t >= 16 * 64 * 32 * 32) return;
    int ow = t % 32;
    int s  = (t / 32) % 32;
    int c  = (t / 1024) % 64;
    int b  = t / (1024 * 64);
    float acc = bias[c];
    const float* wp = w + (size_t)c * 192;
    const float* ip = dnf + (size_t)(b * 64) * 2048 + s * 64;
    int base = 2 * ow - 1;
    for (int ci = 0; ci < 64; ++ci) {
        const float* row = ip + (size_t)ci * 2048;
#pragma unroll
        for (int tt = 0; tt < 3; ++tt) {
            int wi = base + tt;
            if (wi >= 0 && wi < 64) acc += wp[ci * 3 + tt] * row[wi];
        }
    }
    out[65536 + ((size_t)(b * 64 + c)) * 1024 + s * 32 + ow] = gelu_f(bn_f(acc, bn, 64, c));
}

extern "C" void kernel_launch(void* const* d_in, const int* in_sizes, int n_in,
                              void* d_out, int out_size, void* d_ws, size_t ws_size,
                              hipStream_t stream) {
    const float* sparse_fea = (const float*)d_in[0];
    const float* dense_fea  = (const float*)d_in[1];
    const float* dts_w1 = (const float*)d_in[2];
    const float* dts_b1 = (const float*)d_in[3];
    const float* dts_bn1 = (const float*)d_in[4];
    const float* dts_w2 = (const float*)d_in[5];
    const float* dts_b2 = (const float*)d_in[6];
    const float* dts_bn2 = (const float*)d_in[7];
    const float* sp_w1 = (const float*)d_in[8];
    const float* sp_b1 = (const float*)d_in[9];
    const float* sp_bn1 = (const float*)d_in[10];
    const float* sp_w2 = (const float*)d_in[11];
    const float* sp_b2 = (const float*)d_in[12];
    const float* sp_bn2 = (const float*)d_in[13];
    const float* dn_w1 = (const float*)d_in[14];
    const float* dn_b1 = (const float*)d_in[15];
    const float* dn_bn1 = (const float*)d_in[16];
    const float* dn_w2 = (const float*)d_in[17];
    const float* dn_b2 = (const float*)d_in[18];
    const float* dn_bn2 = (const float*)d_in[19];
    const float* smp_w = (const float*)d_in[20];
    const float* smp_b = (const float*)d_in[21];
    const float* smp_bn = (const float*)d_in[22];
    float* out = (float*)d_out;

    uintptr_t base = (uintptr_t)d_ws;
    size_t off = 0;
    auto carve = [&](size_t bytes) -> void* {
        void* p = (void*)(base + off);
        off = (off + bytes + 255) & ~(size_t)255;
        return p;
    };
    float*  h1    = (float*)carve((size_t)16 * 90 * 32 * 63 * 4);
    float*  spdn  = (float*)carve((size_t)16 * 128 * 32 * 4);
    float*  usp   = (float*)carve((size_t)16 * 256 * 32 * 4);
    float*  uden  = (float*)carve((size_t)16 * 192 * 2048 * 4);
    __bf16* xbf   = (__bf16*)carve((size_t)16 * 192 * 2048 * 2);
    float*  sqn   = (float*)carve((size_t)16 * 2048 * 4);
    float*  dist  = (float*)carve((size_t)2048 * 2048 * 4);
    int*    knn   = (int*)carve((size_t)16 * 2048 * 10 * 4);
    float*  Pbuf  = (float*)carve((size_t)16 * 156 * 2048 * 4);
    float*  Qbuf  = (float*)carve((size_t)16 * 156 * 2048 * 4);
    float*  dnf   = (float*)carve((size_t)16 * 64 * 2048 * 4);
    int*    spknn = (int*)carve((size_t)16 * 32 * 2 * 4);

    // DenseToSparse
    {
        int tot = 16 * 90 * 32 * 63;
        k_dts1<<<(tot + 255) / 256, 256, 0, stream>>>(dense_fea, dts_w1, dts_b1, dts_bn1, h1);
    }
    {
        int tot = 16 * 128 * 32;
        k_dts2max<<<(tot + 255) / 256, 256, 0, stream>>>(h1, dts_w2, dts_b2, dts_bn2, spdn);
    }
    // Unions
    {
        int tot = 16 * 256 * 32;
        k_build_usp<<<(tot + 255) / 256, 256, 0, stream>>>(sparse_fea, spdn, usp);
    }
    {
        int tot = 16 * 192 * 2048;
        k_build_uden<<<(tot + 255) / 256, 256, 0, stream>>>(dense_fea, sparse_fea, uden, xbf);
    }
    {
        int tot = 16 * 2048;
        k_sq<<<(tot + 255) / 256, 256, 0, stream>>>(uden, sqn);
    }
    // Sparse branch -> first 65536 elements of d_out
    k_spknn<<<16, 256, 0, stream>>>(usp, spknn);
    k_spdgcnn<<<dim3(32, 16), 256, 0, stream>>>(usp, spknn, sp_w1, sp_b1, sp_bn1,
                                                sp_w2, sp_b2, sp_bn2, out);
    // Dense kNN: per-batch gram+topk (reuse one 2048x2048 dist buffer)
    for (int b = 0; b < 16; ++b) {
        k_gram<<<dim3(32, 128), 128, 0, stream>>>(xbf, sqn, dist, b);
        k_topk<<<256, 256, 0, stream>>>(dist, knn, b);
    }
    // Factored dense conv1: P = Wa X, Q = (Wb - Wa) X
    k_pq<<<dim3(32, 10, 16), 128, 0, stream>>>(dn_w1, xbf, Pbuf, 0);
    k_pq<<<dim3(32, 10, 16), 128, 0, stream>>>(dn_w1, xbf, Qbuf, 1);
    // Fused gather + BN1/GELU + conv2 (WMMA) + BN2/GELU + neighbor-max
    k_dnconv2<<<dim3(128, 16), 128, 0, stream>>>(Pbuf, Qbuf, knn, dn_w2, dn_b1, dn_bn1,
                                                 dn_b2, dn_bn2, dnf);
    // Down-sample -> remaining d_out
    {
        int tot = 16 * 64 * 32 * 32;
        k_down<<<(tot + 255) / 256, 256, 0, stream>>>(dnf, smp_w, smp_b, smp_bn, out);
    }
}
